// Model_21199958573288
// MI455X (gfx1250) — matrix operand order repaired and checked by reading, not compiled, not run on hardware
//
#include <hip/hip_runtime.h>
#include <hip/hip_bf16.h>

typedef __bf16 bf16;
typedef __attribute__((ext_vector_type(16))) __bf16 v16bf;
typedef __attribute__((ext_vector_type(8)))  __bf16 v8bf;
typedef __attribute__((ext_vector_type(8)))  float  v8f;

// ---------------- wave32 reductions ----------------
__device__ __forceinline__ float wave_sum(float v) {
  for (int m = 16; m > 0; m >>= 1) v += __shfl_xor(v, m, 32);
  return v;
}
__device__ __forceinline__ float wave_max(float v) {
  for (int m = 16; m > 0; m >>= 1) v = fmaxf(v, __shfl_xor(v, m, 32));
  return v;
}
__device__ __forceinline__ float gelu_exact(float x) {
  return 0.5f * x * (1.0f + erff(x * 0.70710678118654752440f));
}

// ---------------- WMMA fragment loaders (ISA 7.12.2 layouts) ----------------
// A (16x32, 16-bit): lane = M row (mod 16); half-wave selects K octets
//   VGPR0-3 : K = half*8 .. half*8+7 ;  VGPR4-7 : K = 16+half*8 .. 16+half*8+7
__device__ __forceinline__ v16bf ld_fragA(const bf16* base, int row0, int lane) {
  int r = row0 + (lane & 15);
  int h = (lane >> 4) * 8;
  union { v16bf v; v8bf p[2]; } u;
  u.p[0] = *(const v8bf*)(base + r * 40 + h);
  u.p[1] = *(const v8bf*)(base + r * 40 + 16 + h);
  return u.v;
}
// B (32x16, 16-bit) loaded column-major: lane = N column (mod 16);
//   VGPR0-3 : K = half*8 .. half*8+7 ;  VGPR4-7 : K = 16+half*8 .. 16+half*8+7
__device__ __forceinline__ v16bf ld_fragB(const bf16* base, int row0, int lane) {
  int r = row0 + (lane & 15);
  int h = (lane >> 4) * 8;
  union { v16bf v; v8bf p[2]; } u;
  u.p[0] = *(const v8bf*)(base + r * 40 + h);
  u.p[1] = *(const v8bf*)(base + r * 40 + 16 + h);
  return u.v;
}

// ---------------- WMMA GEMM: C[M,N] = A[M,K] * B[N,K]^T (+C if beta) ----------------
// A,B bf16 row-major; C f32. Batched via z-dim. WG tile 128x128, 8 waves (4Mx2N),
// wave tile 32x64 (8 accumulators), K-step 32, double-buffered LDS (1 barrier/step).
__global__ __launch_bounds__(256) void k_gemm(
    const bf16* __restrict__ A, long long sA, int lda,
    const bf16* __restrict__ B, long long sB, int ldb,
    float* __restrict__ C, long long sC, int ldc,
    int M, int N, int K, int beta)
{
  __shared__ bf16 sAt[2][128 * 40];
  __shared__ bf16 sBt[2][128 * 40];

  const int batch = blockIdx.z;
  A += (long long)batch * sA;
  B += (long long)batch * sB;
  C += (long long)batch * sC;

  const int mBase = blockIdx.y * 128;
  const int nBase = blockIdx.x * 128;
  const int tid  = threadIdx.x;
  const int lane = tid & 31;
  const int wave = tid >> 5;
  const int wm = wave >> 1;          // 0..3  (M)
  const int wn = wave & 1;           // 0..1  (N)

  const int lrow = tid >> 1, lseg = tid & 1;   // 128 rows x 2 halves, 32B/thread

  v8f acc[8] = {};
  uint4 ra0, ra1, rb0, rb1;

  auto gload = [&](int k0) {
    const uint4 z = make_uint4(0u, 0u, 0u, 0u);
    int gr = mBase + lrow;
    ra0 = z; ra1 = z;
    if (gr < M) {
      const uint4* p = (const uint4*)(A + (long long)gr * lda + k0 + lseg * 16);
      ra0 = p[0]; ra1 = p[1];
    }
    int hr = nBase + lrow;
    rb0 = z; rb1 = z;
    if (hr < N) {
      const uint4* p = (const uint4*)(B + (long long)hr * ldb + k0 + lseg * 16);
      rb0 = p[0]; rb1 = p[1];
    }
  };
  auto lstore = [&](int buf) {
    uint4* qa = (uint4*)(sAt[buf] + lrow * 40 + lseg * 16);
    qa[0] = ra0; qa[1] = ra1;
    uint4* qb = (uint4*)(sBt[buf] + lrow * 40 + lseg * 16);
    qb[0] = rb0; qb[1] = rb1;
  };

  gload(0);
  lstore(0);
  __syncthreads();

  int cur = 0;
  for (int k0 = 0; k0 < K; k0 += 32) {
    const bool more = (k0 + 32) < K;
    if (more) gload(k0 + 32);

    v16bf a0 = ld_fragA(sAt[cur], wm * 32,      lane);
    v16bf a1 = ld_fragA(sAt[cur], wm * 32 + 16, lane);
    v16bf b0 = ld_fragB(sBt[cur], wn * 64,      lane);
    v16bf b1 = ld_fragB(sBt[cur], wn * 64 + 16, lane);
    v16bf b2 = ld_fragB(sBt[cur], wn * 64 + 32, lane);
    v16bf b3 = ld_fragB(sBt[cur], wn * 64 + 48, lane);

    acc[0] = __builtin_amdgcn_wmma_f32_16x16x32_bf16(false, a0, false, b0, (short)0, acc[0], false, false);
    acc[1] = __builtin_amdgcn_wmma_f32_16x16x32_bf16(false, a0, false, b1, (short)0, acc[1], false, false);
    acc[2] = __builtin_amdgcn_wmma_f32_16x16x32_bf16(false, a0, false, b2, (short)0, acc[2], false, false);
    acc[3] = __builtin_amdgcn_wmma_f32_16x16x32_bf16(false, a0, false, b3, (short)0, acc[3], false, false);
    acc[4] = __builtin_amdgcn_wmma_f32_16x16x32_bf16(false, a1, false, b0, (short)0, acc[4], false, false);
    acc[5] = __builtin_amdgcn_wmma_f32_16x16x32_bf16(false, a1, false, b1, (short)0, acc[5], false, false);
    acc[6] = __builtin_amdgcn_wmma_f32_16x16x32_bf16(false, a1, false, b2, (short)0, acc[6], false, false);
    acc[7] = __builtin_amdgcn_wmma_f32_16x16x32_bf16(false, a1, false, b3, (short)0, acc[7], false, false);

    if (more) {
      lstore(cur ^ 1);
      __syncthreads();
      cur ^= 1;
    }
  }

  const int half = lane >> 4, colL = lane & 15;
  for (int t = 0; t < 8; ++t) {
    int ti = t >> 2, tj = t & 3;
    int col = nBase + wn * 64 + tj * 16 + colL;
    if (col >= N) continue;
    for (int i = 0; i < 8; ++i) {
      int row = mBase + wm * 32 + ti * 16 + half * 8 + i;
      if (row < M) {
        float* cp = C + (long long)row * ldc + col;
        float v = acc[t][i];
        *cp = beta ? (*cp + v) : v;
      }
    }
  }
}

// ---------------- elementwise / reduction kernels ----------------
__global__ void k_cvt(const float* __restrict__ s, bf16* __restrict__ d, long long n) {
  for (long long i = blockIdx.x * (long long)blockDim.x + threadIdx.x; i < n;
       i += (long long)gridDim.x * blockDim.x)
    d[i] = (bf16)s[i];
}

__global__ void k_zero16(bf16* __restrict__ d, long long n) {
  for (long long i = blockIdx.x * (long long)blockDim.x + threadIdx.x; i < n;
       i += (long long)gridDim.x * blockDim.x)
    d[i] = (bf16)0.0f;
}

// kv[b, r, :] = concat(pt_emb[pts[b,r,0]], pt_emb[pts[b,r,1]])   (bf16)
__global__ void k_build_kv(const int* __restrict__ pts, const float* __restrict__ pt_emb,
                           bf16* __restrict__ kv, long long n) {
  for (long long i = blockIdx.x * (long long)blockDim.x + threadIdx.x; i < n;
       i += (long long)gridDim.x * blockDim.x) {
    int d  = (int)(i % 384);
    long long br = i / 384;
    int half = d / 192, dd = d % 192;
    int tok = pts[br * 2 + half];
    kv[i] = (bf16)pt_emb[(long long)tok * 192 + dd];
  }
}

// h = layernorm(emb[q], norm_w)   (f32 out)
__global__ __launch_bounds__(128) void k_gather_ln(
    const int* __restrict__ q, const float* __restrict__ emb,
    const float* __restrict__ w, float* __restrict__ h)
{
  int wave = threadIdx.x >> 5, lane = threadIdx.x & 31;
  int row = blockIdx.x * 4 + wave;
  if (row >= 12800) return;
  const float* xr = emb + (long long)q[row] * 384;
  float v[12]; float s = 0.0f;
  for (int j = 0; j < 12; ++j) { v[j] = xr[lane + j * 32]; s += v[j]; }
  s = wave_sum(s);
  float mu = s * (1.0f / 384.0f);
  float qq = 0.0f;
  for (int j = 0; j < 12; ++j) { float d = v[j] - mu; qq += d * d; }
  qq = wave_sum(qq);
  float rstd = rsqrtf(qq * (1.0f / 384.0f) + 1e-5f);
  float* orow = h + (long long)row * 384;
  for (int j = 0; j < 12; ++j) { int d = lane + j * 32; orow[d] = (v[j] - mu) * rstd * w[d]; }
}

// out(bf16) = layernorm(x, w)
__global__ __launch_bounds__(128) void k_ln(
    const float* __restrict__ x, const float* __restrict__ w, bf16* __restrict__ out)
{
  int wave = threadIdx.x >> 5, lane = threadIdx.x & 31;
  int row = blockIdx.x * 4 + wave;
  if (row >= 12800) return;
  const float* xr = x + (long long)row * 384;
  float v[12]; float s = 0.0f;
  for (int j = 0; j < 12; ++j) { v[j] = xr[lane + j * 32]; s += v[j]; }
  s = wave_sum(s);
  float mu = s * (1.0f / 384.0f);
  float qq = 0.0f;
  for (int j = 0; j < 12; ++j) { float d = v[j] - mu; qq += d * d; }
  qq = wave_sum(qq);
  float rstd = rsqrtf(qq * (1.0f / 384.0f) + 1e-5f);
  bf16* orow = out + (long long)row * 384;
  for (int j = 0; j < 12; ++j) { int d = lane + j * 32; orow[d] = (bf16)((v[j] - mu) * rstd * w[d]); }
}

// proj row (1632): [0:48]=qr, [48:96]=kr, geglu = lin*gelu(pre);
// geglu[0:384] -> cat[:, 0:384]; geglu[384:768] -> v1T[b, d, s]
__global__ __launch_bounds__(256) void k_split1(
    const float* __restrict__ proj, bf16* __restrict__ qr, bf16* __restrict__ kr,
    bf16* __restrict__ cat, bf16* __restrict__ v1T)
{
  int row = blockIdx.x;                 // 0..12799
  int b = row / 100, s = row % 100;
  const float* pr = proj + (long long)row * 1632;
  for (int c = threadIdx.x; c < 864; c += 256) {
    if (c < 48) {
      qr[(long long)row * 64 + c] = (bf16)pr[c];
    } else if (c < 96) {
      kr[(long long)row * 64 + (c - 48)] = (bf16)pr[c];
    } else {
      int c2 = c - 96;                  // 0..767
      float g = pr[96 + c2] * gelu_exact(pr[864 + c2]);
      if (c2 < 384)
        cat[(long long)row * 768 + c2] = (bf16)g;
      else
        v1T[((long long)b * 384 + (c2 - 384)) * 128 + s] = (bf16)g;
    }
  }
}

// pq row (816): [0:48]=q2, cat[:,0:384] = pq[48:432]*gelu(pq[432:816])
__global__ __launch_bounds__(256) void k_split2(
    const float* __restrict__ pq, bf16* __restrict__ q2, bf16* __restrict__ cat)
{
  int row = blockIdx.x;                 // 0..12799
  const float* pr = pq + (long long)row * 816;
  for (int c = threadIdx.x; c < 432; c += 256) {
    if (c < 48) {
      q2[(long long)row * 64 + c] = (bf16)pr[c];
    } else {
      int c2 = c - 48;
      float g = pr[48 + c2] * gelu_exact(pr[432 + c2]);
      cat[(long long)row * 768 + c2] = (bf16)g;
    }
  }
}

// pk row (816): [0:48]=k2, gvT[b, d, r] = pk[48:432]*gelu(pk[432:816])
__global__ __launch_bounds__(256) void k_split_kv(
    const float* __restrict__ pk, bf16* __restrict__ k2, bf16* __restrict__ gvT)
{
  int row = blockIdx.x;                 // 0..32767
  int b = row >> 8, r = row & 255;
  const float* pr = pk + (long long)row * 816;
  for (int c = threadIdx.x; c < 432; c += 256) {
    if (c < 48) {
      k2[(long long)row * 64 + c] = (bf16)pr[c];
    } else {
      int c2 = c - 48;
      float g = pr[48 + c2] * gelu_exact(pr[432 + c2]);
      gvT[((long long)b * 384 + c2) * 256 + r] = (bf16)g;
    }
  }
}

// causal softmax with softplus(bm)*(j-i) bias; in: (row,128 f32) valid cols<100;
// out bf16 stride 128, pad cols zeroed.
__global__ __launch_bounds__(128) void k_softmax1(
    const float* __restrict__ s, const float* __restrict__ bm, bf16* __restrict__ P)
{
  int wave = threadIdx.x >> 5, lane = threadIdx.x & 31;
  int row = blockIdx.x * 4 + wave;
  if (row >= 12800) return;
  int i = row % 100;
  float sp = log1pf(expf(bm[0]));
  const float scale = 0.14433756729740645f;   // 1/sqrt(48)
  const float* sr = s + (long long)row * 128;
  float v[4]; float m = -3.0e38f;
  for (int jj = 0; jj < 4; ++jj) {
    int j = lane * 4 + jj;
    float val = -3.0e38f;
    if (j < 100) val = sr[j] * scale + ((j <= i) ? sp * (float)(j - i) : -1e30f);
    v[jj] = val; m = fmaxf(m, val);
  }
  m = wave_max(m);
  float sum = 0.0f;
  for (int jj = 0; jj < 4; ++jj) {
    int j = lane * 4 + jj;
    float e = (j < 100) ? expf(v[jj] - m) : 0.0f;
    v[jj] = e; sum += e;
  }
  sum = wave_sum(sum);
  float r = 1.0f / sum;
  bf16* pr = P + (long long)row * 128;
  for (int jj = 0; jj < 4; ++jj) pr[lane * 4 + jj] = (bf16)(v[jj] * r);
}

// plain softmax over 256 cols; out bf16 stride 256
__global__ __launch_bounds__(128) void k_softmax2(
    const float* __restrict__ s, bf16* __restrict__ P)
{
  int wave = threadIdx.x >> 5, lane = threadIdx.x & 31;
  int row = blockIdx.x * 4 + wave;
  if (row >= 12800) return;
  const float scale = 0.14433756729740645f;
  const float* sr = s + (long long)row * 256;
  float v[8]; float m = -3.0e38f;
  for (int jj = 0; jj < 8; ++jj) { v[jj] = sr[lane * 8 + jj] * scale; m = fmaxf(m, v[jj]); }
  m = wave_max(m);
  float sum = 0.0f;
  for (int jj = 0; jj < 8; ++jj) { v[jj] = expf(v[jj] - m); sum += v[jj]; }
  sum = wave_sum(sum);
  float r = 1.0f / sum;
  bf16* pr = P + (long long)row * 256;
  for (int jj = 0; jj < 8; ++jj) pr[lane * 8 + jj] = (bf16)(v[jj] * r);
}

// cat[:, 384:768] = bf16(attn)   attn: (12800,384) f32
__global__ void k_copy_cat(const float* __restrict__ a, bf16* __restrict__ cat) {
  long long n = 12800ll * 384;
  for (long long i = blockIdx.x * (long long)blockDim.x + threadIdx.x; i < n;
       i += (long long)gridDim.x * blockDim.x) {
    long long row = i / 384; int c = (int)(i % 384);
    cat[row * 768 + 384 + c] = (bf16)a[i];
  }
}

// ---------------- host launcher ----------------
extern "C" void kernel_launch(void* const* d_in, const int* in_sizes, int n_in,
                              void* d_out, int out_size, void* d_ws, size_t ws_size,
                              hipStream_t stream) {
  (void)in_sizes; (void)n_in; (void)out_size; (void)ws_size;

  const int*   q_in      = (const int*)d_in[0];
  const int*   pts       = (const int*)d_in[1];
  const float* emb       = (const float*)d_in[2];
  const float* pt_emb    = (const float*)d_in[3];
  const float* norm_w    = (const float*)d_in[4];
  const float* out_w     = (const float*)d_in[5];
  const float* ln1       = (const float*)d_in[6];
  const float* expand    = (const float*)d_in[7];
  const float* project1  = (const float*)d_in[8];
  const float* bias_mult = (const float*)d_in[9];
  const float* ln2       = (const float*)d_in[10];
  const float* Wq        = (const float*)d_in[11];
  const float* Wkv       = (const float*)d_in[12];
  const float* project2  = (const float*)d_in[13];
  float* out = (float*)d_out;

  // ---- workspace carve ----
  char* base = (char*)d_ws;
  size_t off = 0;
  auto carve = [&](size_t bytes) -> char* {
    char* p = base + off;
    off += (bytes + 255) & ~(size_t)255;
    return p;
  };
  float* h    = (float*)carve(12800ull * 384 * 4);
  bf16*  nbuf = (bf16*) carve(12800ull * 384 * 2);
  bf16*  kvb  = (bf16*) carve(32768ull * 384 * 2);
  bf16*  wE   = (bf16*) carve(8ull * 1632 * 384 * 2);
  bf16*  wP1  = (bf16*) carve(8ull * 384 * 768 * 2);
  bf16*  wQw  = (bf16*) carve(8ull * 816 * 384 * 2);
  bf16*  wKVw = (bf16*) carve(8ull * 816 * 384 * 2);
  bf16*  wP2  = (bf16*) carve(8ull * 384 * 768 * 2);
  bf16*  wOut = (bf16*) carve(100ull * 384 * 2);
  float* big  = (float*)carve(32768ull * 816 * 4);
  bf16*  qr   = (bf16*) carve(12800ull * 64 * 2);
  bf16*  kr   = (bf16*) carve(12800ull * 64 * 2);
  bf16*  q2   = (bf16*) carve(12800ull * 64 * 2);
  bf16*  k2   = (bf16*) carve(32768ull * 64 * 2);
  bf16*  cat  = (bf16*) carve(12800ull * 768 * 2);
  bf16*  v1T  = (bf16*) carve(128ull * 384 * 128 * 2);
  bf16*  gvT  = (bf16*) carve(128ull * 384 * 256 * 2);
  float* sc   = (float*)carve(12800ull * 256 * 4);
  bf16*  Pb   = (bf16*) carve(12800ull * 256 * 2);
  float* attn = (float*)carve(12800ull * 384 * 4);

  auto gs = [](long long n) -> unsigned {
    long long b = (n + 255) / 256;
    return (unsigned)(b > 4096 ? 4096 : b);
  };
  auto gemm = [&](const bf16* A, long long sA, int lda,
                  const bf16* Bp, long long sB, int ldb,
                  float* Cp, long long sC, int ldc,
                  int M, int N, int K, int beta, int batches) {
    dim3 g((unsigned)((N + 127) / 128), (unsigned)((M + 127) / 128), (unsigned)batches);
    k_gemm<<<g, 256, 0, stream>>>(A, sA, lda, Bp, sB, ldb, Cp, sC, ldc, M, N, K, beta);
  };

  // ---- setup: weight conversion, pad zeroing, kv gather, input LN ----
  k_cvt<<<gs(8ll*1632*384), 256, 0, stream>>>(expand,   wE,   8ll*1632*384);
  k_cvt<<<gs(8ll*384*768),  256, 0, stream>>>(project1, wP1,  8ll*384*768);
  k_cvt<<<gs(8ll*816*384),  256, 0, stream>>>(Wq,       wQw,  8ll*816*384);
  k_cvt<<<gs(8ll*816*384),  256, 0, stream>>>(Wkv,      wKVw, 8ll*816*384);
  k_cvt<<<gs(8ll*384*768),  256, 0, stream>>>(project2, wP2,  8ll*384*768);
  k_cvt<<<gs(100ll*384),    256, 0, stream>>>(out_w,    wOut, 100ll*384);

  k_zero16<<<gs(12800ll*64),      256, 0, stream>>>(qr,  12800ll*64);
  k_zero16<<<gs(12800ll*64),      256, 0, stream>>>(kr,  12800ll*64);
  k_zero16<<<gs(12800ll*64),      256, 0, stream>>>(q2,  12800ll*64);
  k_zero16<<<gs(32768ll*64),      256, 0, stream>>>(k2,  32768ll*64);
  k_zero16<<<gs(128ll*384*128),   256, 0, stream>>>(v1T, 128ll*384*128);

  k_build_kv<<<gs(32768ll*384), 256, 0, stream>>>(pts, pt_emb, kvb, 32768ll*384);
  k_gather_ln<<<3200, 128, 0, stream>>>(q_in, emb, norm_w, h);

  // ---- 8 sequential layers ----
  for (int l = 0; l < 8; ++l) {
    // n1 = LN(h, ln1[l]); proj = n1 @ We^T
    k_ln<<<3200, 128, 0, stream>>>(h, ln1 + l * 384, nbuf);
    gemm(nbuf, 0, 384, wE + (long long)l * 1632 * 384, 0, 384,
         big, 0, 1632, 12800, 1632, 384, 0, 1);
    k_split1<<<12800, 256, 0, stream>>>(big, qr, kr, cat, v1T);

    // attention 1 (causal, biased)
    gemm(qr, 100ll*64, 64, kr, 100ll*64, 64, sc, 100ll*128, 128, 100, 100, 64, 0, 128);
    k_softmax1<<<3200, 128, 0, stream>>>(sc, bias_mult + l, Pb);
    gemm(Pb, 100ll*128, 128, v1T, 384ll*128, 128, attn, 100ll*384, 384, 100, 384, 128, 0, 128);
    k_copy_cat<<<gs(12800ll*384), 256, 0, stream>>>(attn, cat);

    // h += cat @ Wp1^T
    gemm(cat, 0, 768, wP1 + (long long)l * 384 * 768, 0, 768,
         h, 0, 384, 12800, 384, 768, 1, 1);

    // n2 = LN(h, ln2[l]); pq = n2 @ Wq^T
    k_ln<<<3200, 128, 0, stream>>>(h, ln2 + l * 384, nbuf);
    gemm(nbuf, 0, 384, wQw + (long long)l * 816 * 384, 0, 384,
         big, 0, 816, 12800, 816, 384, 0, 1);
    k_split2<<<12800, 256, 0, stream>>>(big, q2, cat);

    // pk = kv @ Wkv^T
    gemm(kvb, 0, 384, wKVw + (long long)l * 816 * 384, 0, 384,
         big, 0, 816, 32768, 816, 384, 0, 1);
    k_split_kv<<<32768, 256, 0, stream>>>(big, k2, gvT);

    // attention 2 (cross, plain)
    gemm(q2, 100ll*64, 64, k2, 256ll*64, 64, sc, 100ll*256, 256, 100, 256, 64, 0, 128);
    k_softmax2<<<3200, 128, 0, stream>>>(sc, Pb);
    gemm(Pb, 100ll*256, 256, gvT, 384ll*256, 256, attn, 100ll*384, 384, 100, 384, 256, 0, 128);
    k_copy_cat<<<gs(12800ll*384), 256, 0, stream>>>(attn, cat);

    // h += cat @ Wp2^T
    gemm(cat, 0, 768, wP2 + (long long)l * 384 * 768, 0, 768,
         h, 0, 384, 12800, 384, 768, 1, 1);
  }

  // ---- final LN + output projection ----
  k_ln<<<3200, 128, 0, stream>>>(h, norm_w, nbuf);
  gemm(nbuf, 0, 384, wOut, 0, 384, out, 0, 100, 12800, 100, 384, 0, 1);
}
